// DelayedXOR_Vanilla_SNN_2637109920460
// MI455X (gfx1250) — compile-verified, matrix operand order checked by reading, not run on hardware
//
#include <hip/hip_runtime.h>
#include <hip/hip_bf16.h>

typedef __bf16 bf16_t;
typedef __attribute__((ext_vector_type(16))) __bf16 v16bf;
typedef __attribute__((ext_vector_type(8)))  __bf16 v8bf;
typedef __attribute__((ext_vector_type(8)))  float  v8f;

union V16U { v16bf v; v8bf h[2]; };

#define B_ 128
#define T_ 2048
#define I_ 128
#define H_ 1024
#define O_ 128
#define NWG 64
#define TAU_INV 0.5f

// ---------------------------------------------------------------------------
// Grid-wide sense-style barrier (monotonic generation counter in L2).
// Only thread 0 of each WG participates in the atomics; everyone else parks
// on __syncthreads(). One barrier per timestep is the only global sync.
// ---------------------------------------------------------------------------
__device__ __forceinline__ void grid_barrier(unsigned* count, unsigned* gen) {
  __syncthreads();
  if (threadIdx.x == 0) {
    __threadfence();                       // release our s1 stores
    volatile unsigned* vg = gen;
    const unsigned old = *vg;              // read generation BEFORE arriving
    const unsigned arrived = atomicAdd(count, 1u);
    if (arrived == NWG - 1) {
      atomicExch(count, 0u);
      __threadfence();
      atomicAdd(gen, 1u);                  // open the gate
    } else {
      while (*vg == old) { __builtin_amdgcn_s_sleep(1); }
    }
    __threadfence();                       // acquire everyone else's stores
  }
  __syncthreads();
}

// ---------------------------------------------------------------------------
// One-shot helpers: bf16 weight images + zeroed state/accumulator/barrier.
// Re-run every kernel_launch call -> deterministic across graph replays.
// ---------------------------------------------------------------------------
__global__ void convert_weights(const float* __restrict__ W1,
                                const float* __restrict__ W2,
                                bf16_t* __restrict__ W1b,
                                bf16_t* __restrict__ W2b) {
  const int i = blockIdx.x * blockDim.x + threadIdx.x;
  if (i < H_ * I_) W1b[i] = (__bf16)W1[i];
  if (i < O_ * H_) W2b[i] = (__bf16)W2[i];
}

__global__ void init_state(float* __restrict__ v1, float* __restrict__ v2,
                           float* __restrict__ out, unsigned* __restrict__ bar) {
  const int i = blockIdx.x * blockDim.x + threadIdx.x;
  if (i < B_ * H_) v1[i] = 0.0f;
  if (i < B_ * O_) { v2[i] = 0.0f; out[i] = 0.0f; }
  if (i < 2) bar[i] = 0u;
}

// ---------------------------------------------------------------------------
// Persistent SNN kernel: 64 WGs x 128 threads (4 waves, wave32).
// Phase A: layer-1 GEMM via v_wmma_f32_16x16x32_bf16 (512 tiles / 256 waves),
//          LIF on WG-owned v1 slice, spikes -> double-buffered bf16 s1.
// grid_barrier (the single per-step sync).
// Phase B: layer-2 GEMM, K=1024 split over the 4 waves (8 WMMA each),
//          LDS reduce, LIF on WG-owned v2 tile, accumulate output (t>=T/2).
// ---------------------------------------------------------------------------
__global__ __launch_bounds__(128) void snn_persistent(
    const float* __restrict__ x,     // [B][T][I]
    const float* __restrict__ b1,    // [H]
    const float* __restrict__ b2,    // [O]
    const bf16_t* __restrict__ W1b,  // [H][I]
    const bf16_t* __restrict__ W2b,  // [O][H]
    bf16_t* __restrict__ s1buf,      // 2 x [B][H]
    float* __restrict__ v1,          // [B][H]
    float* __restrict__ v2,          // [B][O]
    float* __restrict__ out,         // [B][O]
    unsigned* __restrict__ bar)      // {count, gen}
{
  const int wg    = blockIdx.x;          // 0..63
  const int wave  = threadIdx.x >> 5;    // 0..3
  const int lane  = threadIdx.x & 31;
  const int lr    = lane & 15;           // row (A) / col (B,C) within 16x16
  const int khalf = (lane >> 4) << 3;    // K lo/hi split per ISA A/B layout

  __shared__ float red[4][32][8];        // cross-wave K-reduction, 4 KB

  const int tr2 = wg >> 3;               // layer-2 tile (fixed per WG)
  const int tc2 = wg & 7;
  const int gw  = wg * 4 + wave;         // global wave id 0..255

  for (int t = 0; t < T_; ++t) {
    bf16_t* s1 = s1buf + (size_t)(t & 1) * (B_ * H_);

    // ---------------- Phase A: layer 1 ----------------
#pragma unroll
    for (int rep = 0; rep < 2; ++rep) {
      const int tile = gw + rep * 256;   // 0..511 = 8 row-tiles x 64 col-tiles
      const int tr   = tile >> 6;
      const int tc   = tile & 63;
      const int arow = tr * 16 + lr;
      const int bcol = tc * 16 + lr;
      const float*  ap = x   + ((size_t)arow * T_ + t) * I_;  // x[b][t][:]
      const bf16_t* bp = W1b + (size_t)bcol * I_;             // W1 row = B col
      v8f acc = {};
#pragma unroll
      for (int k = 0; k < I_; k += 32) {
        const int klo = k + khalf;
        const int khi = klo + 16;
        const v8f alo = *(const v8f*)(ap + klo);   // 2x b128 f32
        const v8f ahi = *(const v8f*)(ap + khi);
        V16U A, Bm;
#pragma unroll
        for (int j = 0; j < 8; ++j) {
          A.v[j]     = (__bf16)alo[j];             // f32 -> bf16 on the fly
          A.v[j + 8] = (__bf16)ahi[j];
        }
        Bm.h[0] = *(const v8bf*)(bp + klo);        // contiguous bf16 weights
        Bm.h[1] = *(const v8bf*)(bp + khi);
        acc = __builtin_amdgcn_wmma_f32_16x16x32_bf16(
            false, A.v, false, Bm.v, (short)0, acc, false, false);
      }
      if (t + 1 < T_) __builtin_prefetch(ap + I_, 0, 1);  // next-step x slice
      const float bias = b1[bcol];
#pragma unroll
      for (int j = 0; j < 8; ++j) {                // C layout: M = 16tr+khalf+j
        const int    row = tr * 16 + khalf + j;
        const size_t vi  = (size_t)row * H_ + bcol;
        const float  h   = acc[j] + bias;
        float v = v1[vi];
        v += (h - v) * TAU_INV;                    // v += (h - v)/tau
        const bool fire = (v >= 1.0f);
        v1[vi] = fire ? 0.0f : v;                  // hard reset
        s1[vi] = fire ? (__bf16)1.0f : (__bf16)0.0f;
      }
    }

    grid_barrier(bar, bar + 1);   // all s1[t&1] visible; one sync per step

    // ---------------- Phase B: layer 2 ----------------
    {
      const int arow = tr2 * 16 + lr;
      const int bcol = tc2 * 16 + lr;
      const bf16_t* ap = s1  + (size_t)arow * H_;
      const bf16_t* bp = W2b + (size_t)bcol * H_;
      v8f acc = {};
      const int k0 = wave * 256;                   // K split across 4 waves
#pragma unroll 4
      for (int k = k0; k < k0 + 256; k += 32) {
        const int klo = k + khalf;
        const int khi = klo + 16;
        V16U A, Bm;
        A.h[0]  = *(const v8bf*)(ap + klo);
        A.h[1]  = *(const v8bf*)(ap + khi);
        Bm.h[0] = *(const v8bf*)(bp + klo);
        Bm.h[1] = *(const v8bf*)(bp + khi);
        acc = __builtin_amdgcn_wmma_f32_16x16x32_bf16(
            false, A.v, false, Bm.v, (short)0, acc, false, false);
      }
#pragma unroll
      for (int j = 0; j < 8; ++j) red[wave][lane][j] = acc[j];
      __syncthreads();
      if (wave == 0) {
        const float bias = b2[bcol];
#pragma unroll
        for (int j = 0; j < 8; ++j) {
          const float h = red[0][lane][j] + red[1][lane][j] +
                          red[2][lane][j] + red[3][lane][j] + bias;
          const int    row = tr2 * 16 + khalf + j;
          const size_t vi  = (size_t)row * O_ + bcol;
          float v = v2[vi];
          v += (h - v) * TAU_INV;
          const bool fire = (v >= 1.0f);
          v2[vi] = fire ? 0.0f : v;
          if (t >= T_ / 2 && fire) out[vi] += 1.0f;   // sum of s2 over 2nd half
        }
      }
      // red[] reuse is fenced by the __syncthreads inside next grid_barrier
    }
  }
}

// ---------------------------------------------------------------------------
extern "C" void kernel_launch(void* const* d_in, const int* in_sizes, int n_in,
                              void* d_out, int out_size, void* d_ws, size_t ws_size,
                              hipStream_t stream) {
  const float* x  = (const float*)d_in[0];   // [128][2048][128] f32
  const float* W1 = (const float*)d_in[1];   // [1024][128] f32
  const float* b1 = (const float*)d_in[2];   // [1024] f32
  const float* W2 = (const float*)d_in[3];   // [128][1024] f32
  const float* b2 = (const float*)d_in[4];   // [128] f32
  float* out = (float*)d_out;                // [128][128] f32

  char* ws = (char*)d_ws;                    // ~1.6 MB total, L2-resident
  bf16_t*   W1b = (bf16_t*)(ws);                      // 256 KB
  bf16_t*   W2b = (bf16_t*)(ws +  256 * 1024);        // 256 KB
  bf16_t*   s1  = (bf16_t*)(ws +  512 * 1024);        // 2x256 KB (double buf)
  float*    v1  = (float*) (ws + 1024 * 1024);        // 512 KB
  float*    v2  = (float*) (ws + 1536 * 1024);        //  64 KB
  unsigned* bar = (unsigned*)(ws + 1600 * 1024);      // {count, gen}

  convert_weights<<<512, 256, 0, stream>>>(W1, W2, W1b, W2b);
  init_state<<<512, 256, 0, stream>>>(v1, v2, out, bar);
  snn_persistent<<<NWG, 128, 0, stream>>>(x, b1, b2, W1b, W2b,
                                          s1, v1, v2, out, bar);
}
